// XBOXAttention16_69518340653744
// MI455X (gfx1250) — matrix operand order repaired and checked by reading, not compiled, not run on hardware
//
#include <hip/hip_runtime.h>

// ---------------------------------------------------------------------------
// XBOX attention bucketing for MI455X (gfx1250, wave32, WMMA)
// B=1, H=16, S=2048, D=64, bucket=32
// ---------------------------------------------------------------------------

typedef __attribute__((ext_vector_type(8)))  _Float16 v8h;
typedef __attribute__((ext_vector_type(16))) _Float16 v16h;
typedef __attribute__((ext_vector_type(8)))  float    v8f;

#define H_    16
#define S_    2048
#define D_    64
#define DP_   65        // D+1
#define KP_   96        // K padded to 3 x 32 for wmma f16 k=32
#define TOPK_ 32

static __device__ __forceinline__ float hrnd(float f) {            // round to fp16, back to f32
  return (float)(_Float16)f;
}
static __device__ __forceinline__ unsigned short f2hb(float f) {   // float -> fp16 bits
  _Float16 h = (_Float16)f; unsigned short u; __builtin_memcpy(&u, &h, 2); return u;
}

// argmax over one 8-element fp16 group, skipping masked elements.
// Strict '>' with ascending e keeps the lowest index on ties.
static __device__ __forceinline__ void group_argmax(v8h c, unsigned mbyte,
                                                    float& gv, int& ge) {
  gv = -__builtin_inff(); ge = 8;            // ge==8 -> empty group
  #pragma unroll
  for (int e = 0; e < 8; ++e) {
    const float v = (float)c[e];
    if (!((mbyte >> e) & 1u) && v > gv) { gv = v; ge = e; }
  }
}

// combine 8 group maxima into the lane's best (val, global j), min-j ties.
static __device__ __forceinline__ void lane_best(const float* gv, const int* ge,
                                                 int lane, float& lv, int& lj) {
  lv = -__builtin_inff(); lj = 0x7FFFFFFF;
  #pragma unroll
  for (int g = 0; g < 8; ++g) {
    const int j = (ge[g] < 8) ? (g * 256 + lane * 8 + ge[g]) : 0x7FFFFFFF;
    if (gv[g] > lv || (gv[g] == lv && j < lj)) { lv = gv[g]; lj = j; }
  }
}

// ---------------------------------------------------------------------------
// Kernel 0: init scalar slots in workspace (ws is poisoned by harness)
// ---------------------------------------------------------------------------
__global__ void k_init(unsigned int* phi_bits) {
  if (threadIdx.x == 0) *phi_bits = 0u;
}

// ---------------------------------------------------------------------------
// Kernel 1: per-row fp16 norms of qk, global max (phi) via uint atomicMax
// One wave per (h,s) row; 8 waves / block.
// ---------------------------------------------------------------------------
__global__ __launch_bounds__(256) void k_norms(const float* __restrict__ qk,
                                               float* __restrict__ norms,
                                               unsigned int* __restrict__ phi_bits) {
  const int wave = threadIdx.x >> 5;
  const int lane = threadIdx.x & 31;
  const int hs   = blockIdx.x * 8 + wave;
  const float* row = qk + (size_t)hs * D_;
  float v0 = hrnd(row[lane]);
  float v1 = hrnd(row[lane + 32]);
  float s = v0 * v0 + v1 * v1;
  #pragma unroll
  for (int off = 16; off; off >>= 1) s += __shfl_xor(s, off, 32);
  if (lane == 0) {
    float n = sqrtf(s);
    norms[hs] = n;
    atomicMax(phi_bits, __float_as_uint(n));   // n >= 0 -> uint order == float order
  }
}

// ---------------------------------------------------------------------------
// Kernel 2: build fp16 P_pad[hs][96], a_pad[hs][96], and Qs[hs] (fp32).
//   P = [qk, sqrt(phi^2 - n^2)] * (phi / ||P||)   (analytically scale ~= 1)
//   Q_proj = (phi/n) * dot(qk, a[0:64])           (Q's last component is 0)
// One wave per row.
// ---------------------------------------------------------------------------
__global__ __launch_bounds__(256) void k_build(const float* __restrict__ qk,
                                               const float* __restrict__ a,
                                               const float* __restrict__ norms,
                                               const unsigned int* __restrict__ phi_bits,
                                               _Float16* __restrict__ Ppad,
                                               _Float16* __restrict__ Apad,
                                               float* __restrict__ Qs) {
  const int wave = threadIdx.x >> 5;
  const int lane = threadIdx.x & 31;
  const int hs   = blockIdx.x * 8 + wave;

  const float phi = __uint_as_float(*phi_bits);
  const float n   = norms[hs];
  const float c   = sqrtf(fmaxf(phi * phi - n * n, 0.0f));
  const float pn  = sqrtf(n * n + c * c);
  const float ps  = (pn > 0.0f) ? (phi / pn) : 0.0f;   // P scale (~1)
  const float qs  = (n  > 0.0f) ? (phi / n ) : 0.0f;   // Q scale

  const float*   qrow = qk + (size_t)hs * D_;
  const float*   arow = a  + (size_t)hs * DP_;
  _Float16*      prow = Ppad + (size_t)hs * KP_;
  _Float16*      xrow = Apad + (size_t)hs * KP_;

  const float q0 = hrnd(qrow[lane]);
  const float q1 = hrnd(qrow[lane + 32]);
  const float a0 = hrnd(arow[lane]);
  const float a1 = hrnd(arow[lane + 32]);

  prow[lane]      = (_Float16)(q0 * ps);
  prow[lane + 32] = (_Float16)(q1 * ps);
  prow[lane + 64] = (lane == 0) ? (_Float16)(c * ps) : (_Float16)0.0f;

  xrow[lane]      = (_Float16)a0;
  xrow[lane + 32] = (_Float16)a1;
  xrow[lane + 64] = (lane == 0) ? (_Float16)arow[64] : (_Float16)0.0f;

  float d = q0 * qs * a0 + q1 * qs * a1;
  #pragma unroll
  for (int off = 16; off; off >>= 1) d += __shfl_xor(d, off, 32);
  if (lane == 0) Qs[hs] = hrnd(d);
}

// ---------------------------------------------------------------------------
// Kernel 3: per (head, 16-row tile):
//   scores[16][2048] = (a_pad @ P_pad^T) * Qs[j] + mask[j]  via v_wmma
//   -> LDS (fp16), register-resident top-32 per row, stream {0,-10000} fp32.
// Block = 256 threads (8 waves). grid = (S/16, H).
// ---------------------------------------------------------------------------
__global__ __launch_bounds__(256) void k_score(const _Float16* __restrict__ Apad,
                                               const _Float16* __restrict__ Ppad,
                                               const float* __restrict__ Qs,
                                               const float* __restrict__ mask,
                                               float* __restrict__ out) {
  __shared__ unsigned short sm[16][S_];          // 64 KB fp16 score tile

  const int h     = blockIdx.y;
  const int i0    = blockIdx.x * 16;
  const int wave  = threadIdx.x >> 5;
  const int lane  = threadIdx.x & 31;
  const int lhalf = lane >> 4;                   // 0: lanes 0-15, 1: lanes 16-31
  const int l16   = lane & 15;

  // --- A fragments: rows i0..i0+15, three K chunks (reused across j tiles) ---
  // A 16x32 fp16 layout: lanes 0-15 row M=lane, K = {k0..k0+7, k0+16..k0+23};
  //                      lanes 16-31 row M=lane-16, K = {k0+8..k0+15, k0+24..k0+31}.
  v16h Afrag[3];
  {
    const _Float16* abase = Apad + ((size_t)(h * S_ + i0 + l16)) * KP_;
    #pragma unroll
    for (int kc = 0; kc < 3; ++kc) {
      const int koff = kc * 32 + lhalf * 8;
      v8h lo = *(const v8h*)(abase + koff);
      v8h hi = *(const v8h*)(abase + koff + 16);
      v16h A;
      #pragma unroll
      for (int t = 0; t < 8; ++t) { A[t] = lo[t]; A[t + 8] = hi[t]; }
      Afrag[kc] = A;
    }
  }

  // --- sweep j tiles: wave w handles tiles w, w+8, ... ---
  for (int jt = wave; jt < S_ / 16; jt += 8) {
    const int j0 = jt * 16;
    // B 32x16 fp16 layout: lane holds column N=lane&15 (= P row j0+N),
    // K = {k0+8h..k0+8h+7, k0+16+8h..k0+16+8h+7}, h = lane>>4 (matches A).
    const _Float16* pbase = Ppad + ((size_t)(h * S_ + j0 + l16)) * KP_ + lhalf * 8;
    __builtin_prefetch(pbase + 8 * 16 * KP_, 0, 0);   // next tile for this wave

    v8f acc = {};
    #pragma unroll
    for (int kc = 0; kc < 3; ++kc) {
      v8h lo = *(const v8h*)(pbase + kc * 32);
      v8h hi = *(const v8h*)(pbase + kc * 32 + 16);
      v16h B;
      #pragma unroll
      for (int t = 0; t < 8; ++t) { B[t] = lo[t]; B[t + 8] = hi[t]; }
      acc = __builtin_amdgcn_wmma_f32_16x16x32_f16(
          /*neg_a=*/false, Afrag[kc], /*neg_b=*/false, B,
          /*c_mod=*/(short)0, acc, /*reuse_a=*/false, /*reuse_b=*/false);
    }

    // epilogue: D tile lane layout: VGPR v -> row v + 8*lhalf, col = l16
    const int j  = j0 + l16;
    const float q = Qs[h * S_ + j];
    const float m = mask[j];
    #pragma unroll
    for (int v = 0; v < 8; ++v) {
      float sc = acc[v] * q;
      sc = (sc != sc) ? 0.0f : sc;     // jnp.where(isnan, 0)
      sc += m;
      sm[v + lhalf * 8][j] = f2hb(sc);
    }
  }
  __syncthreads();

  // --- register-resident top-32; wave w owns rows 2w, 2w+1 ---
  // Lane owns j = t*256 + lane*8 + e  (t,e in 0..7): conflict-free b128 LDS
  // loads, ascending (t,e) == ascending j for top_k's lowest-index tie-break.
  for (int rr = 0; rr < 2; ++rr) {
    const int r = wave * 2 + rr;

    v8h ch[8];
    #pragma unroll
    for (int t = 0; t < 8; ++t)
      ch[t] = *(const v8h*)&sm[r][t * 256 + lane * 8];

    unsigned long long selmask = 0ull;           // bit g*8+e = selected
    float gv[8]; int ge[8];
    #pragma unroll
    for (int t = 0; t < 8; ++t) group_argmax(ch[t], 0u, gv[t], ge[t]);
    float lv; int lj;
    lane_best(gv, ge, lane, lv, lj);

    for (int it = 0; it < TOPK_; ++it) {
      float bv = lv; int bj = lj;
      #pragma unroll
      for (int off = 16; off; off >>= 1) {
        const float ov = __shfl_xor(bv, off, 32);
        const int   oj = __shfl_xor(bj, off, 32);
        if (ov > bv || (ov == bv && oj < bj)) { bv = ov; bj = oj; }
      }
      // winner lane (unique: lane id is embedded in j bits [7:3]) updates
      if (bj < S_ && ((bj >> 3) & 31) == lane) {
        const int g = bj >> 8;
        selmask |= 1ull << (((unsigned)g << 3) | (unsigned)(bj & 7));
        switch (g) {
          case 0: group_argmax(ch[0], (unsigned)(selmask >>  0) & 0xFFu, gv[0], ge[0]); break;
          case 1: group_argmax(ch[1], (unsigned)(selmask >>  8) & 0xFFu, gv[1], ge[1]); break;
          case 2: group_argmax(ch[2], (unsigned)(selmask >> 16) & 0xFFu, gv[2], ge[2]); break;
          case 3: group_argmax(ch[3], (unsigned)(selmask >> 24) & 0xFFu, gv[3], ge[3]); break;
          case 4: group_argmax(ch[4], (unsigned)(selmask >> 32) & 0xFFu, gv[4], ge[4]); break;
          case 5: group_argmax(ch[5], (unsigned)(selmask >> 40) & 0xFFu, gv[5], ge[5]); break;
          case 6: group_argmax(ch[6], (unsigned)(selmask >> 48) & 0xFFu, gv[6], ge[6]); break;
          default: group_argmax(ch[7], (unsigned)(selmask >> 56) & 0xFFu, gv[7], ge[7]); break;
        }
        lane_best(gv, ge, lane, lv, lj);
      }
    }

    // stream the row straight from the selection mask (no LDS re-read):
    // per t, lanes cover 1 KB contiguously (32 lanes x 32 B).
    const size_t obase = ((size_t)h * S_ + (size_t)(i0 + r)) * (size_t)S_;
    #pragma unroll
    for (int t = 0; t < 8; ++t) {
      const unsigned m8 = (unsigned)(selmask >> (t * 8)) & 0xFFu;
      const int j0 = t * 256 + lane * 8;
      float4 o0, o1;
      o0.x = (m8 & 0x01u) ? 0.0f : -10000.0f;
      o0.y = (m8 & 0x02u) ? 0.0f : -10000.0f;
      o0.z = (m8 & 0x04u) ? 0.0f : -10000.0f;
      o0.w = (m8 & 0x08u) ? 0.0f : -10000.0f;
      o1.x = (m8 & 0x10u) ? 0.0f : -10000.0f;
      o1.y = (m8 & 0x20u) ? 0.0f : -10000.0f;
      o1.z = (m8 & 0x40u) ? 0.0f : -10000.0f;
      o1.w = (m8 & 0x80u) ? 0.0f : -10000.0f;
      *(float4*)(out + obase + j0)     = o0;
      *(float4*)(out + obase + j0 + 4) = o1;
    }
  }
}

// ---------------------------------------------------------------------------
// Launcher
// ---------------------------------------------------------------------------
extern "C" void kernel_launch(void* const* d_in, const int* in_sizes, int n_in,
                              void* d_out, int out_size, void* d_ws, size_t ws_size,
                              hipStream_t stream) {
  (void)in_sizes; (void)n_in; (void)out_size; (void)ws_size;

  const float* qk   = (const float*)d_in[0];   // [16,2048,64]  f32
  const float* mask = (const float*)d_in[1];   // [2048]        f32
  const float* a    = (const float*)d_in[2];   // [16,2048,65]  f32
  float*       out  = (float*)d_out;           // [16,2048,2048] f32

  // workspace layout (16B aligned slices)
  char* w = (char*)d_ws;
  unsigned int* phi_bits = (unsigned int*)w;                          // 4 B (256 B slot)
  float*        norms    = (float*)(w + 256);                         // 128 KB
  float*        Qs       = (float*)(w + 256 + 131072);                // 128 KB
  _Float16*     Ppad     = (_Float16*)(w + 256 + 262144);             // 6 MB
  _Float16*     Apad     = (_Float16*)(w + 256 + 262144 + 6291456);   // 6 MB

  const int rows = H_ * S_;                    // 32768 rows, 8 rows (waves) / block

  k_init <<<1, 1, 0, stream>>>(phi_bits);
  k_norms<<<rows / 8, 256, 0, stream>>>(qk, norms, phi_bits);
  k_build<<<rows / 8, 256, 0, stream>>>(qk, a, norms, phi_bits, Ppad, Apad, Qs);
  k_score<<<dim3(S_ / 16, H_), 256, 0, stream>>>(Apad, Ppad, Qs, mask, out);
}
